// GATNodeEdgePredictionWithFlowIntensity_4243427688597
// MI455X (gfx1250) — matrix operand order repaired and checked by reading, not compiled, not run on hardware
//
#include <hip/hip_runtime.h>
#include <math.h>

// Problem constants (from reference)
#define NN   50000
#define EE   800000
#define IND  128
#define EDD  16
#define HH1  4
#define CC1  32
#define OUTD 64

typedef __attribute__((ext_vector_type(16))) __bf16 v16bf;
typedef __attribute__((ext_vector_type(8)))  float  v8f;

struct B32x2 { uint4 x, y; };

__device__ __forceinline__ unsigned short f2bf_bits(float f) {
    unsigned int u = __float_as_uint(f);
    unsigned int r = (u + 0x7FFFu + ((u >> 16) & 1u)) >> 16;
    return (unsigned short)r;
}

__device__ __forceinline__ void atomicMaxF(float* addr, float val) {
    if (val >= 0.f) atomicMax((int*)addr, __float_as_int(val));
    else            atomicMin((unsigned int*)addr, (unsigned int)__float_as_int(val));
}

// ---------------- generic small kernels ----------------

__global__ void k_fill(float* __restrict__ p, int n, float v) {
    int t = blockIdx.x * blockDim.x + threadIdx.x;
    if (t < n) p[t] = v;
}

__global__ void k_f2bf(const float* __restrict__ x, unsigned short* __restrict__ y, int n) {
    int t = blockIdx.x * blockDim.x + threadIdx.x;
    if (t < n) y[t] = f2bf_bits(x[t]);
}

// W [K][Nc] f32 row-major -> Wt [Nc][K] bf16 (so B-fragment loads are contiguous)
__global__ void k_prepw(const float* __restrict__ W, unsigned short* __restrict__ Wt, int K, int Nc) {
    int t = blockIdx.x * blockDim.x + threadIdx.x;
    if (t >= K * Nc) return;
    int n = t / K, k = t % K;
    Wt[t] = f2bf_bits(W[(size_t)k * Nc + n]);
}

// wevec[h*16+k] = sum_c We[k, h*C+c] * ae[h*C+c]
__global__ void k_wevec(const float* __restrict__ We, const float* __restrict__ ae,
                        float* __restrict__ wevec, int Hh, int C) {
    int t = threadIdx.x;
    if (t >= Hh * 16) return;
    int h = t / 16, k = t % 16;
    int Nc = Hh * C;
    float acc = 0.f;
    for (int c = 0; c < C; ++c) acc += We[(size_t)k * Nc + h * C + c] * ae[h * C + c];
    wevec[t] = acc;
}

// ---------------- self-loop mean edge_attr ----------------

__global__ void k_degsum(const int* __restrict__ dst, const float* __restrict__ ea,
                         float* __restrict__ deg, float* __restrict__ sum, int E_) {
    int e = blockIdx.x * blockDim.x + threadIdx.x;
    if (e >= E_) return;
    int di = dst[e];
    atomicAdd(&deg[di], 1.f);
    const float* row = ea + (size_t)e * EDD;
    float* srow = sum + (size_t)di * EDD;
    #pragma unroll
    for (int k = 0; k < EDD; ++k) atomicAdd(&srow[k], row[k]);
}

__global__ void k_mean(float* __restrict__ sum, const float* __restrict__ deg, int Nn) {
    int t = blockIdx.x * blockDim.x + threadIdx.x;
    if (t >= Nn * EDD) return;
    sum[t] = sum[t] / fmaxf(deg[t / EDD], 1.f);
}

// ---------------- WMMA GEMM: C[M x Nc] = A_bf16[M x K] * Wt_bf16[Nc x K]^T ----------------
// one wave per 16x16 output tile; K in steps of 32 via v_wmma_f32_16x16x32_bf16

__global__ __launch_bounds__(128)
void k_gemm_bf16(const unsigned short* __restrict__ Au, const unsigned short* __restrict__ Bu,
                 float* __restrict__ Cm, int Mtiles, int Ntiles, int K) {
    const __bf16* A  = (const __bf16*)Au;
    const __bf16* Bt = (const __bf16*)Bu;
    int wave = blockIdx.x * (blockDim.x >> 5) + (threadIdx.x >> 5);
    int lane = threadIdx.x & 31;
    if (wave >= Mtiles * Ntiles) return;
    int tm = wave / Ntiles, tn = wave % Ntiles;
    int hi = lane >> 4;
    int row = tm * 16 + (lane & 15);
    int col = tn * 16 + (lane & 15);

    // A fragment (16x32 bf16): lane row=l&15; elems 0..7 = K[kh..kh+7], 8..15 = K[kh+16..kh+23], kh = k0 + 8*(l>=16)
    const __bf16* arow = A  + (size_t)row * K + hi * 8;
    // B fragment (32x16): lane col=l&15; elems 0..7 = K[kh..kh+7], 8..15 = K[kh+16..kh+23], kh = k0 + 8*(l>=16), from transposed W
    const __bf16* brow = Bt + (size_t)col * K + hi * 8;

    v8f c = {};
    for (int k0 = 0; k0 < K; k0 += 32) {
        B32x2 ab;
        ab.x = *(const uint4*)(arow + k0);
        ab.y = *(const uint4*)(arow + k0 + 16);
        v16bf a = __builtin_bit_cast(v16bf, ab);
        B32x2 bb;
        bb.x = *(const uint4*)(brow + k0);
        bb.y = *(const uint4*)(brow + k0 + 16);
        v16bf b = __builtin_bit_cast(v16bf, bb);
        c = __builtin_amdgcn_wmma_f32_16x16x32_bf16(false, a, false, b, (short)0, c, false, false);
    }

    // D layout: VGPR r -> row (r + 8*hi), col = lane&15
    int Ncols = Ntiles * 16;
    float* out = Cm + (size_t)(tm * 16 + hi * 8) * Ncols + col;
    #pragma unroll
    for (int r = 0; r < 8; ++r) out[(size_t)r * Ncols] = c[r];
}

// ---------------- GAT per-layer pieces ----------------

// per-node attention scores s/d: s[n,h] = xp[n, hC:(h+1)C] . a_s[h]
__global__ void k_scores(const float* __restrict__ xp, const float* __restrict__ as_,
                         const float* __restrict__ ad_, float* __restrict__ s,
                         float* __restrict__ d, int Nn, int Hh, int C) {
    int t = blockIdx.x * blockDim.x + threadIdx.x;
    if (t >= Nn * Hh) return;
    int n = t / Hh, h = t % Hh;
    const float* row = xp + (size_t)n * (Hh * C) + h * C;
    float ss = 0.f, dd = 0.f;
    for (int c = 0; c < C; ++c) { float v = row[c]; ss += v * as_[h * C + c]; dd += v * ad_[h * C + c]; }
    s[t] = ss; d[t] = dd;
}

// edge logits (leaky 0.2) + atomic segment-max over dst; e in [E, E+N) are self-loops
__global__ void k_logits(const int* __restrict__ src, const int* __restrict__ dst,
                         const float* __restrict__ ea, const float* __restrict__ meanattr,
                         const float* __restrict__ s, const float* __restrict__ dsc,
                         const float* __restrict__ wevec, float* __restrict__ logit,
                         float* __restrict__ m, int E_, int N_, int Hh) {
    int e = blockIdx.x * blockDim.x + threadIdx.x;
    if (e >= E_ + N_) return;
    int si, di; const float* eav;
    if (e < E_) { si = src[e]; di = dst[e]; eav = ea + (size_t)e * EDD; }
    else        { si = di = e - E_;          eav = meanattr + (size_t)(e - E_) * EDD; }
    float ev[EDD];
    #pragma unroll
    for (int k = 0; k < EDD; ++k) ev[k] = eav[k];
    for (int h = 0; h < Hh; ++h) {
        float l = s[si * Hh + h] + dsc[di * Hh + h];
        #pragma unroll
        for (int k = 0; k < EDD; ++k) l += ev[k] * wevec[h * 16 + k];
        l = l > 0.f ? l : 0.2f * l;
        logit[(size_t)e * Hh + h] = l;
        atomicMaxF(&m[di * Hh + h], l);
    }
}

// ex = exp(logit - m[dst]); den[dst] += ex  (logit buffer overwritten in place)
__global__ void k_exp(const int* __restrict__ dst, float* __restrict__ logit,
                      const float* __restrict__ m, float* __restrict__ den,
                      int E_, int N_, int Hh) {
    int t = blockIdx.x * blockDim.x + threadIdx.x;
    if (t >= (E_ + N_) * Hh) return;
    int e = t / Hh, h = t % Hh;
    int di = e < E_ ? dst[e] : e - E_;
    float ex = __expf(logit[t] - m[di * Hh + h]);
    logit[t] = ex;
    atomicAdd(&den[di * Hh + h], ex);
}

// aggregation: one wave per edge; lane handles 4 channels (float4 gather + 4 atomic adds)
__global__ __launch_bounds__(256)
void k_agg(const int* __restrict__ src, const int* __restrict__ dst,
           const float* __restrict__ ex, const float* __restrict__ den,
           const float* __restrict__ xp, float* __restrict__ agg,
           int E_, int N_, int Hh, int C, int D) {
    int lane = threadIdx.x & 31;
    int e = blockIdx.x * (blockDim.x >> 5) + (threadIdx.x >> 5);
    if (e >= E_ + N_) return;
    int c0 = lane * 4;
    if (c0 >= D) return;
    int si, di;
    if (e < E_) { si = src[e]; di = dst[e]; }
    else        { si = di = e - E_; }
    int h = c0 / C;  // C is a multiple of 4 => all 4 channels share one head
    float alpha = ex[(size_t)e * Hh + h] / (den[di * Hh + h] + 1e-16f);
    float4 v = *(const float4*)(xp + (size_t)si * D + c0);
    float* ap = agg + (size_t)di * D + c0;
    atomicAdd(ap + 0, v.x * alpha);
    atomicAdd(ap + 1, v.y * alpha);
    atomicAdd(ap + 2, v.z * alpha);
    atomicAdd(ap + 3, v.w * alpha);
}

// z = leaky_relu(agg + b, 0.01)
__global__ void k_final(const float* __restrict__ agg, const float* __restrict__ b,
                        float* __restrict__ z, int Nn, int D) {
    int t = blockIdx.x * blockDim.x + threadIdx.x;
    if (t >= Nn * D) return;
    float v = agg[t] + b[t % D];
    z[t] = v > 0.f ? v : 0.01f * v;
}

// ---------------- host orchestration ----------------

static inline unsigned blks(long n, int bs = 256) { return (unsigned)((n + bs - 1) / bs); }

extern "C" void kernel_launch(void* const* d_in, const int* in_sizes, int n_in,
                              void* d_out, int out_size, void* d_ws, size_t ws_size,
                              hipStream_t stream) {
    (void)in_sizes; (void)n_in; (void)out_size; (void)ws_size;

    const float* x    = (const float*)d_in[0];
    const int*   ei   = (const int*)d_in[1];
    const float* ea   = (const float*)d_in[2];
    const float* W1   = (const float*)d_in[3];
    const float* We1  = (const float*)d_in[4];
    const float* as1  = (const float*)d_in[5];
    const float* ad1  = (const float*)d_in[6];
    const float* ae1  = (const float*)d_in[7];
    const float* b1   = (const float*)d_in[8];
    const float* W2   = (const float*)d_in[9];
    const float* We2  = (const float*)d_in[10];
    const float* as2  = (const float*)d_in[11];
    const float* ad2  = (const float*)d_in[12];
    const float* ae2  = (const float*)d_in[13];
    const float* b2   = (const float*)d_in[14];
    const float* W3   = (const float*)d_in[15];
    const float* We3  = (const float*)d_in[16];
    const float* as3  = (const float*)d_in[17];
    const float* ad3  = (const float*)d_in[18];
    const float* ae3  = (const float*)d_in[19];
    const float* b3   = (const float*)d_in[20];

    const int* srcp = ei;
    const int* dstp = ei + EE;

    // workspace carve-up (256B aligned)
    char* ws = (char*)d_ws;
    size_t off = 0;
    auto alloc = [&](size_t bytes) -> void* {
        void* p = ws + off;
        off = (off + bytes + 255) & ~(size_t)255;
        return p;
    };
    float*          deg      = (float*)alloc((size_t)NN * 4);
    float*          meanattr = (float*)alloc((size_t)NN * EDD * 4);
    unsigned short* xbf      = (unsigned short*)alloc((size_t)NN * IND * 2);
    unsigned short* wt       = (unsigned short*)alloc((size_t)IND * IND * 2);
    float*          wevec    = (float*)alloc(64 * 4);
    float*          xp       = (float*)alloc((size_t)NN * IND * 4);
    float*          zbuf     = (float*)alloc((size_t)NN * IND * 4);
    float*          sbuf     = (float*)alloc((size_t)NN * HH1 * 4);
    float*          dbuf     = (float*)alloc((size_t)NN * HH1 * 4);
    float*          mbuf     = (float*)alloc((size_t)NN * HH1 * 4);
    float*          denbuf   = (float*)alloc((size_t)NN * HH1 * 4);
    float*          exbuf    = (float*)alloc((size_t)(EE + NN) * HH1 * 4);
    float*          agg      = (float*)alloc((size_t)NN * IND * 4);

    // self-loop mean edge attr
    k_fill<<<blks(NN), 256, 0, stream>>>(deg, NN, 0.f);
    k_fill<<<blks((long)NN * EDD), 256, 0, stream>>>(meanattr, NN * EDD, 0.f);
    k_degsum<<<blks(EE), 256, 0, stream>>>(dstp, ea, deg, meanattr, EE);
    k_mean<<<blks((long)NN * EDD), 256, 0, stream>>>(meanattr, deg, NN);

    const int Mtiles = NN / 16;  // 3125, exact
    const int Etot = EE + NN;

    auto layer = [&](const float* fin, const float* W, const float* We, const float* as_,
                     const float* ad_, const float* ae_, const float* bb,
                     int Hh, int C, float* zout) {
        const int Dout = Hh * C;
        const int Ntiles = Dout / 16;
        // operand prep
        k_f2bf<<<blks((long)NN * IND), 256, 0, stream>>>(fin, xbf, NN * IND);
        k_prepw<<<blks((long)IND * Dout), 256, 0, stream>>>(W, wt, IND, Dout);
        k_wevec<<<1, 64, 0, stream>>>(We, ae_, wevec, Hh, C);
        // node GEMM via WMMA bf16 (fp32 accumulate)
        k_gemm_bf16<<<blks((long)Mtiles * Ntiles, 4) , 128, 0, stream>>>(xbf, wt, xp, Mtiles, Ntiles, IND);
        // per-node scores
        k_scores<<<blks((long)NN * Hh), 256, 0, stream>>>(xp, as_, ad_, sbuf, dbuf, NN, Hh, C);
        // softmax state init
        k_fill<<<blks((long)NN * Hh), 256, 0, stream>>>(mbuf, NN * Hh, -INFINITY);
        k_fill<<<blks((long)NN * Hh), 256, 0, stream>>>(denbuf, NN * Hh, 0.f);
        k_fill<<<blks((long)NN * Dout), 256, 0, stream>>>(agg, NN * Dout, 0.f);
        // edge softmax + aggregation
        k_logits<<<blks(Etot), 256, 0, stream>>>(srcp, dstp, ea, meanattr, sbuf, dbuf,
                                                 wevec, exbuf, mbuf, EE, NN, Hh);
        k_exp<<<blks((long)Etot * Hh), 256, 0, stream>>>(dstp, exbuf, mbuf, denbuf, EE, NN, Hh);
        k_agg<<<blks((long)Etot * 32, 256), 256, 0, stream>>>(srcp, dstp, exbuf, denbuf, xp, agg,
                                                              EE, NN, Hh, C, Dout);
        k_final<<<blks((long)NN * Dout), 256, 0, stream>>>(agg, bb, zout, NN, Dout);
    };

    layer(x,    W1, We1, as1, ad1, ae1, b1, HH1, CC1, zbuf);
    layer(zbuf, W2, We2, as2, ad2, ae2, b2, HH1, CC1, zbuf);
    layer(zbuf, W3, We3, as3, ad3, ae3, b3, 1, OUTD, (float*)d_out);
}
